// TimeAwareSlidingWindowPyramidAttention_87943750353394
// MI455X (gfx1250) — compile-verified, matrix operand order checked by reading, not run on hardware
//
#include <hip/hip_runtime.h>

typedef __attribute__((ext_vector_type(16))) _Float16 v16h;
typedef __attribute__((ext_vector_type(8)))  float    v8f;

// ---------------------------------------------------------------------------
// CDNA5 v_wmma_f32_16x16x32_f16 (wave32) fragment layout (ISA 7.12.2):
//   A 16x32: lane holds row M=lane&15; half h covers
//            K = (h<8?0:16) + (lane&16?8:0) + (h&7)
//   B 32x16: lane holds col N=lane&15; same K mapping
//   C/D v8f: reg r, lane: M = r + (lane&16?8:0), N = lane&15
// We store A and B pre-swizzled as 32-lane records of 16 contiguous halves
// (32B/lane), so a fragment load is one v16h load (2x b128).
// ---------------------------------------------------------------------------
__device__ __forceinline__ int wmma_kidx(int lane, int h) {
  return ((h < 8) ? 0 : 16) + ((lane & 16) ? 8 : 0) + (h & 7);
}

// scatter index for weight matrices: record = (m>>4)*KB + (k>>5)
__device__ __forceinline__ int swzA_idx(int m, int k, int KB) {
  int k5 = k & 31;
  int lane = ((k5 >> 3) & 1) * 16 + (m & 15);
  int h = ((k5 >> 4) & 1) * 8 + (k5 & 7);
  return (((m >> 4) * KB + (k >> 5)) * 32 + lane) * 16 + h;
}

// scatter index for activation tiles (N = 64 px -> NB = 4): record = (k>>5)*4 + (n>>4)
__device__ __forceinline__ int swzB_idx(int k, int n) {
  int k5 = k & 31;
  int lane = ((k5 >> 3) & 1) * 16 + (n & 15);
  int h = ((k5 >> 4) & 1) * 8 + (k5 & 7);
  return (((k >> 5) * 4 + (n >> 4)) * 32 + lane) * 16 + h;
}

__device__ __forceinline__ v16h load_frag(const _Float16* base, int rec, int lane) {
  return *(const v16h*)(base + (size_t)rec * 512 + lane * 16);
}

__device__ __forceinline__ v8f wmma16(v16h a, v16h b, v8f c) {
  return __builtin_amdgcn_wmma_f32_16x16x32_f16(false, a, false, b, (short)0, c, false, false);
}

__device__ __forceinline__ void store_d_lds(float* O, int ldn, int m0, int n0, int lane,
                                            v8f d, const float* bias) {
  int n = n0 + (lane & 15);
  int mb = m0 + ((lane & 16) ? 8 : 0);
#pragma unroll
  for (int r = 0; r < 8; ++r) O[(mb + r) * ldn + n] = d[r] + bias[mb + r];
}

__device__ __forceinline__ void pack2_lds(_Float16* dst, int idx, float v0, float v1) {
  union { _Float16 h[2]; unsigned u; } pk;
  pk.h[0] = (_Float16)v0; pk.h[1] = (_Float16)v1;
  *(unsigned*)(dst + idx) = pk.u;   // idx even -> dword aligned
}

// ---------------------------------------------------------------------------
// K0: weights -> f16, pre-swizzled into WMMA A-fragment layout; FiLM; |dt|; hscale
// f16 ws layout (halves): so1@0, aw1@8192, so2@16384, aw2@22528,
//                         vproj@25600, oproj@41984, conf1@58368 (end 95232)
// ---------------------------------------------------------------------------
__global__ void prep_kernel(
    const float* so1_w, const float* aw1_w, const float* so2_w, const float* aw2_w,
    const float* vproj_w, const float* oproj_w, const float* conf1_w,
    const float* tgt, const float* filmo_w, const float* filmo_b,
    const float* filma_w, const float* filma_b,
    const float* rel, const float* dts_w, const float* dts_b,
    _Float16* wh, float* film, float* absdt, float* hscale)
{
  int tid = blockIdx.x * blockDim.x + threadIdx.x;
  if (tid < 95232) {
    float val; int idx;
    if (tid < 8192)       { int off = tid;         val = so1_w[off];   idx = 0     + swzA_idx(off >> 7, off & 127, 4); }
    else if (tid < 16384) { int off = tid - 8192;  val = aw1_w[off];   idx = 8192  + swzA_idx(off >> 7, off & 127, 4); }
    else if (tid < 22528) { int off = tid - 16384; val = so2_w[off];   idx = 16384 + swzA_idx(off >> 6, off & 63, 2); }
    else if (tid < 25600) { int off = tid - 22528; val = aw2_w[off];   idx = 22528 + swzA_idx(off >> 6, off & 63, 2); }
    else if (tid < 41984) { int off = tid - 25600; val = vproj_w[off]; idx = 25600 + swzA_idx(off >> 7, off & 127, 4); }
    else if (tid < 58368) { int off = tid - 41984; val = oproj_w[off]; idx = 41984 + swzA_idx(off >> 7, off & 127, 4); }
    else                  { int off = tid - 58368; int m = off / 1152, k = off - m * 1152;
                            val = conf1_w[off];    idx = 58368 + swzA_idx(m, k, 36); }
    wh[idx] = (_Float16)val;
  } else if (tid < 96256) {
    int idx = tid - 95232;
    int which = idx >> 9; int r = idx & 511;
    int b = r >> 7; int c = r & 127;
    const float* Wm = which ? filma_w : filmo_w;
    const float* Bb = which ? filma_b : filmo_b;
    float s = Bb[c];
    for (int k = 0; k < 64; ++k) s += tgt[b * 64 + k] * Wm[c * 64 + k];
    film[idx] = s;
  } else if (tid < 96268) {
    int idx = tid - 96256;
    absdt[idx] = fabsf(rel[idx]);
  } else if (tid < 96316) {
    int idx = tid - 96268;               // (b*3+n)*4+h
    int b = idx / 12; int h = idx & 3; int n = (idx >> 2) - b * 3;
    float ad = fabsf(rel[b * 3 + n]);
    float z = ad * dts_w[h] + dts_b[h];
    hscale[idx] = 1.0f + 0.5f / (1.0f + expf(-z));   // 1 + (MOS-1)*sigmoid
  }
}

// ---------------------------------------------------------------------------
// K1: per-window offsets/attention branches + softmax + entropy + coords
// ---------------------------------------------------------------------------
__global__ __launch_bounds__(256) void branch_kernel(
    const float* __restrict__ query,
    const _Float16* __restrict__ wh,
    const float* __restrict__ film,
    const float* __restrict__ absdt,
    const float* __restrict__ hscale,
    const float* so1_b, const float* gno_g, const float* gno_b, const float* so2_b,
    const float* aw1_b, const float* gna_g, const float* gna_b, const float* aw2_b,
    float* __restrict__ attnW, float* __restrict__ coordW,
    float* __restrict__ entOut)
{
  __shared__ _Float16 __attribute__((aligned(32))) sX[128 * 64];  // swizzled B; first 4096 reused as X2
  __shared__ float sH[64 * 64];         // hidden (plain); reused as attn_raw (48x64)
  __shared__ float sOff[96 * 64];       // offsets (plain)
  __shared__ float sStats[16];
  __shared__ float sEnt[64];

  const int t = threadIdx.x, lane = t & 31, wv = t >> 5;
  const int bw = blockIdx.x;
  const int b  = bw >> 8;
  const int wi = (bw >> 4) & 15, wj = bw & 15;
  const int fb = bw & 3;                // reference's tile(): index = bw % B

  if (t < 64) sEnt[t] = 0.0f;

  const _Float16* so1h = wh;
  const _Float16* aw1h = wh + 8192;
  const _Float16* so2h = wh + 16384;
  const _Float16* aw2h = wh + 22528;

  for (int br = 0; br < 2; ++br) {
    const float* filmv = film + (br ? 512 : 0) + fb * 128;
    const _Float16* w1 = br ? aw1h : so1h;
    const _Float16* w2 = br ? aw2h : so2h;
    const float* bb1 = br ? aw1_b : so1_b;
    const float* bb2 = br ? aw2_b : so2_b;
    const float* gg  = br ? gna_g : gno_g;
    const float* gb  = br ? gna_b : gno_b;
    float* out2 = br ? sH : sOff;

    // load q window + film, write swizzled f16 (channel pairs -> one b32 store)
#pragma unroll
    for (int e = 0; e < 16; ++e) {
      int pi = e * 256 + t;
      int c0 = (pi >> 6) * 2, px = pi & 63;
      int i = px >> 3, j = px & 7;
      size_t g0 = (((size_t)b * 128 + c0) * 128 + (wi * 8 + i)) * 128 + (wj * 8 + j);
      float v0 = query[g0] + filmv[c0];
      float v1 = query[g0 + 16384] + filmv[c0 + 1];
      pack2_lds(sX, swzB_idx(c0, px), v0, v1);
    }
    __syncthreads();

    // GEMM1: h = W1(64x128) @ X(128x64) + b1
    {
      v8f acc0 = {}; v8f acc1 = {};
      int t0 = wv * 2, t1 = t0 + 1;
      int mt0 = t0 >> 2, nt0 = t0 & 3, mt1 = t1 >> 2, nt1 = t1 & 3;
#pragma unroll
      for (int ks = 0; ks < 4; ++ks) {
        v16h a0 = load_frag(w1, mt0 * 4 + ks, lane);
        v16h bf0 = load_frag(sX, ks * 4 + nt0, lane);
        acc0 = wmma16(a0, bf0, acc0);
        v16h a1 = load_frag(w1, mt1 * 4 + ks, lane);
        v16h bf1 = load_frag(sX, ks * 4 + nt1, lane);
        acc1 = wmma16(a1, bf1, acc1);
      }
      store_d_lds(sH, 64, mt0 * 16, nt0 * 16, lane, acc0, bb1);
      store_d_lds(sH, 64, mt1 * 16, nt1 * 16, lane, acc1, bb1);
    }
    __syncthreads();

    // group norm stats: wave wv handles group wv (8 ch x 64 px = 512 vals)
    {
      float s = 0.f, sq = 0.f;
      int base = wv * 512 + lane * 16;
#pragma unroll
      for (int e = 0; e < 16; ++e) { float v = sH[base + e]; s += v; sq += v * v; }
#pragma unroll
      for (int m = 16; m >= 1; m >>= 1) { s += __shfl_xor(s, m, 32); sq += __shfl_xor(sq, m, 32); }
      if (lane == 0) {
        float mu = s * (1.0f / 512.0f);
        float var = sq * (1.0f / 512.0f) - mu * mu;
        sStats[wv * 2] = mu;
        sStats[wv * 2 + 1] = rsqrtf(var + 1e-5f);
      }
    }
    __syncthreads();

    // normalize + affine + relu -> swizzled f16 X2 (reuse sX)
    _Float16* sX2 = sX;
#pragma unroll
    for (int e = 0; e < 8; ++e) {
      int pi = e * 256 + t;                 // 2048 channel-pairs
      int c0 = (pi >> 6) * 2, px = pi & 63;
      int g = c0 >> 3;
      float mu = sStats[g * 2], rs = sStats[g * 2 + 1];
      float x0 = fmaxf((sH[c0 * 64 + px] - mu) * rs * gg[c0] + gb[c0], 0.0f);
      float x1 = fmaxf((sH[(c0 + 1) * 64 + px] - mu) * rs * gg[c0 + 1] + gb[c0 + 1], 0.0f);
      pack2_lds(sX2, swzB_idx(c0, px), x0, x1);
    }
    __syncthreads();

    // GEMM2: out = W2(M x 64) @ X2(64x64) + b2   (M=96 offsets / 48 attn)
    {
      int ntiles = (br ? 3 : 6) * 4;
      for (int ti = wv; ti < ntiles; ti += 8) {
        int mt = ti >> 2, nt = ti & 3;
        v8f acc = {};
#pragma unroll
        for (int ks = 0; ks < 2; ++ks) {
          v16h a = load_frag(w2, mt * 2 + ks, lane);
          v16h bf = load_frag(sX2, ks * 4 + nt, lane);
          acc = wmma16(a, bf, acc);
        }
        store_d_lds(out2, 64, mt * 16, nt * 16, lane, acc, bb2);
      }
    }
    __syncthreads();
  }

  // postprocess: softmax over (l,p)=12 per (px,head), entropy, sample coords
  {
    int px = t & 63, h = t >> 6;
    int i = px >> 3, j = px & 7;
    float logits[12];
    float mx = -1e30f;
#pragma unroll
    for (int l = 0; l < 3; ++l) {
      float bias = -absdt[fb * 3 + l];
#pragma unroll
      for (int p = 0; p < 4; ++p) {
        float z = sH[(((h * 3 + l) * 4 + p)) * 64 + px] + bias;
        logits[l * 4 + p] = z;
        mx = fmaxf(mx, z);
      }
    }
    float ssum = 0.f;
#pragma unroll
    for (int k = 0; k < 12; ++k) { logits[k] = expf(logits[k] - mx); ssum += logits[k]; }
    float inv = 1.0f / ssum;
    float enth = 0.f;
    size_t abase = ((size_t)(bw * 64 + px) * 4 + h) * 12;
#pragma unroll
    for (int k = 0; k < 12; ++k) {
      float a = logits[k] * inv;
      attnW[abase + k] = a;
      float pr = a + 1e-8f;
      enth -= pr * logf(pr);
    }
    atomicAdd(&sEnt[px], enth * 0.25f);     // mean over 4 heads

    float refx = (float)j * (1.0f / 7.0f);
    float refy = (float)i * (1.0f / 7.0f);
#pragma unroll
    for (int l = 0; l < 3; ++l) {
      float hs = hscale[(b * 3 + l) * 4 + h];   // repeat(): index = bw / num_win
#pragma unroll
      for (int p = 0; p < 4; ++p) {
        int ch = ((h * 3 + l) * 4 + p) * 2;
        float x = fminf(fmaxf(refx + sOff[(ch + 0) * 64 + px], 0.0f), 1.0f) * 2.0f - 1.0f;
        float y = fminf(fmaxf(refy + sOff[(ch + 1) * 64 + px], 0.0f), 1.0f) * 2.0f - 1.0f;
        x = fminf(fmaxf(x * hs, -1.0f), 1.0f);
        y = fminf(fmaxf(y * hs, -1.0f), 1.0f);
        float ix = fminf(fmaxf(((x + 1.0f) * 8.0f - 1.0f) * 0.5f, 0.0f), 7.0f);
        float iy = fminf(fmaxf(((y + 1.0f) * 8.0f - 1.0f) * 0.5f, 0.0f), 7.0f);
        size_t cbase = (abase + l * 4 + p) * 2;
        coordW[cbase + 0] = ix;
        coordW[cbase + 1] = iy;
      }
    }
  }
  __syncthreads();
  if (t < 64) {
    int i = t >> 3, j = t & 7;
    entOut[((size_t)b * 128 + (wi * 8 + i)) * 128 + (wj * 8 + j)] = sEnt[t];
  }
}

// ---------------------------------------------------------------------------
// K2: v_proj = vproj_w @ (v + level_embed) + b  -> f16 ws  [n][bw][c][px] (plain)
// ---------------------------------------------------------------------------
__global__ __launch_bounds__(256) void vproj_kernel(
    const float* __restrict__ values, const float* __restrict__ level_embed,
    const _Float16* __restrict__ wh, const float* __restrict__ vproj_b,
    _Float16* __restrict__ vout)
{
  __shared__ _Float16 __attribute__((aligned(32))) sX[128 * 64];
  const int t = threadIdx.x, lane = t & 31, wv = t >> 5;
  const int gid = blockIdx.x;
  const int n = gid >> 10, bw = gid & 1023;
  const int b = bw >> 8, wi = (bw >> 4) & 15, wj = bw & 15;
  const _Float16* vprojh = wh + 25600;

#pragma unroll
  for (int e = 0; e < 16; ++e) {
    int pi = e * 256 + t;
    int c0 = (pi >> 6) * 2, px = pi & 63;
    int i = px >> 3, j = px & 7;
    size_t g0 = ((((size_t)b * 3 + n) * 128 + c0) * 128 + (wi * 8 + i)) * 128 + (wj * 8 + j);
    float v0 = values[g0] + level_embed[n * 128 + c0];
    float v1 = values[g0 + 16384] + level_embed[n * 128 + c0 + 1];
    pack2_lds(sX, swzB_idx(c0, px), v0, v1);
  }
  __syncthreads();

  size_t obase = (size_t)gid * 8192;
#pragma unroll
  for (int tt = 0; tt < 4; ++tt) {
    int ti = wv * 4 + tt;
    int mt = ti >> 2, nt = ti & 3;
    v8f acc = {};
#pragma unroll
    for (int ks = 0; ks < 4; ++ks) {
      v16h a = load_frag(vprojh, mt * 4 + ks, lane);
      v16h bf = load_frag(sX, ks * 4 + nt, lane);
      acc = wmma16(a, bf, acc);
    }
    int nn = nt * 16 + (lane & 15);
    int mb = mt * 16 + ((lane & 16) ? 8 : 0);
#pragma unroll
    for (int r = 0; r < 8; ++r) {
      int c = mb + r;
      vout[obase + (size_t)c * 64 + nn] = (_Float16)(acc[r] + vproj_b[c]);
    }
  }
}

// ---------------------------------------------------------------------------
// K3: async-DMA v tiles to LDS, bilinear sample + attention-weighted sum,
//     oproj GEMM, window-reversed fused output
// ---------------------------------------------------------------------------
__global__ __launch_bounds__(256) void sample_oproj_kernel(
    const _Float16* __restrict__ vout,
    const float* __restrict__ attnW, const float* __restrict__ coordW,
    const _Float16* __restrict__ wh, const float* __restrict__ oproj_b,
    float* __restrict__ fused)
{
  __shared__ float sAcc[128 * 64];
  __shared__ _Float16 __attribute__((aligned(32))) sV[128 * 64];  // plain v tile; reused swizzled for oproj
  const int t = threadIdx.x, lane = t & 31, wv = t >> 5;
  const int bw = blockIdx.x;
  const int b = bw >> 8, wi = (bw >> 4) & 15, wj = bw & 15;
  const _Float16* oprojh = wh + 41984;

#pragma unroll
  for (int e = 0; e < 32; ++e) sAcc[e * 256 + t] = 0.0f;

  for (int n = 0; n < 3; ++n) {
    __syncthreads();
    size_t vbase = ((size_t)n * 1024 + bw) * 8192;
    // CDNA5 async DMA: global -> LDS, tracked by ASYNCcnt
    {
      unsigned ldsbase = (unsigned)(unsigned long long)(&sV[0]);
#pragma unroll
      for (int e = 0; e < 4; ++e) {
        int idx = (e * 256 + t) * 8;               // halves; 16B per op
        unsigned laddr = ldsbase + (unsigned)(idx * 2);
        const _Float16* ga = vout + vbase + idx;
        asm volatile("global_load_async_to_lds_b128 %0, %1, off"
                     :: "v"(laddr), "v"(ga) : "memory");
      }
      asm volatile("s_wait_asynccnt 0x0" ::: "memory");
    }
    if (n < 2) __builtin_prefetch(vout + vbase + 8192 + t * 32, 0, 1);
    __syncthreads();

#pragma unroll
    for (int e = 0; e < 32; ++e) {
      int lin = e * 256 + t;
      int c = lin >> 6, px = lin & 63, h = c >> 5;
      size_t abase = ((size_t)(bw * 64 + px) * 4 + h) * 12 + n * 4;
      const _Float16* vc = sV + c * 64;
      float acc = 0.0f;
#pragma unroll
      for (int p = 0; p < 4; ++p) {
        float wgt = attnW[abase + p];
        float ix = coordW[(abase + p) * 2 + 0];
        float iy = coordW[(abase + p) * 2 + 1];
        float x0f = floorf(ix), y0f = floorf(iy);
        int x0 = (int)x0f, y0 = (int)y0f;
        int x1 = min(x0 + 1, 7), y1 = min(y0 + 1, 7);
        float wx = ix - x0f, wy = iy - y0f;
        float v00 = (float)vc[y0 * 8 + x0], v01 = (float)vc[y0 * 8 + x1];
        float v10 = (float)vc[y1 * 8 + x0], v11 = (float)vc[y1 * 8 + x1];
        float top = v00 + (v01 - v00) * wx;
        float bot = v10 + (v11 - v10) * wx;
        acc += wgt * (top + (bot - top) * wy);
      }
      sAcc[lin] += acc;
    }
  }
  __syncthreads();
  // convert accumulators to swizzled f16 B-tile (reuse sV)
#pragma unroll
  for (int e = 0; e < 16; ++e) {
    int pi = e * 256 + t;
    int c0 = (pi >> 6) * 2, px = pi & 63;
    pack2_lds(sV, swzB_idx(c0, px), sAcc[c0 * 64 + px], sAcc[(c0 + 1) * 64 + px]);
  }
  __syncthreads();

#pragma unroll
  for (int tt = 0; tt < 4; ++tt) {
    int ti = wv * 4 + tt;
    int mt = ti >> 2, nt = ti & 3;
    v8f acc = {};
#pragma unroll
    for (int ks = 0; ks < 4; ++ks) {
      v16h a = load_frag(oprojh, mt * 4 + ks, lane);
      v16h bf = load_frag(sV, ks * 4 + nt, lane);
      acc = wmma16(a, bf, acc);
    }
    int px = nt * 16 + (lane & 15);
    int i = px >> 3, j = px & 7;
    int mb = mt * 16 + ((lane & 16) ? 8 : 0);
#pragma unroll
    for (int r = 0; r < 8; ++r) {
      int c = mb + r;
      fused[(((size_t)b * 128 + c) * 128 + (wi * 8 + i)) * 128 + (wj * 8 + j)] = acc[r] + oproj_b[c];
    }
  }
}

// ---------------------------------------------------------------------------
// K4: conf = sigmoid(conf2 @ relu(conv3x3(fused))) -- im2col WMMA, K=1152
// ---------------------------------------------------------------------------
__global__ __launch_bounds__(256) void conf_kernel(
    const float* __restrict__ fused,
    const _Float16* __restrict__ wh,
    const float* __restrict__ conf1_b,
    const float* __restrict__ conf2_w, const float* __restrict__ conf2_b,
    float* __restrict__ confOut)
{
  __shared__ _Float16 sIn[128 * 100];     // [c][10][10] halo tile
  __shared__ float sC1[32 * 64];
  const int t = threadIdx.x, lane = t & 31, wv = t >> 5;
  const int bid = blockIdx.x;
  const int b = bid >> 8, ti_ = (bid >> 4) & 15, tj_ = bid & 15;
  const _Float16* conf1h = wh + 58368;    // swizzled A: 32 x 1152, KB=36

  for (int e = 0; e < 50; ++e) {
    int lin = e * 256 + t;
    int c = lin / 100, r = lin % 100;
    int dy = r / 10, dx = r % 10;
    int gy = ti_ * 8 + dy - 1, gx = tj_ * 8 + dx - 1;
    float v = 0.0f;
    if (gy >= 0 && gy < 128 && gx >= 0 && gx < 128)
      v = fused[(((size_t)b * 128 + c) * 128 + gy) * 128 + gx];
    sIn[lin] = (_Float16)v;
  }
  __syncthreads();

  int mt = wv >> 2, nt = wv & 3;          // one 16x16 tile per wave (2x4 tiles)
  int col = nt * 16 + (lane & 15);
  int pi = col >> 3, pj = col & 7;
  v8f acc = {};
  for (int ks = 0; ks < 36; ++ks) {
    int k0 = ks * 32;
    v16h a = load_frag(conf1h, mt * 36 + ks, lane);
    v16h bf;
#pragma unroll
    for (int hh = 0; hh < 16; ++hh) {
      int k = k0 + wmma_kidx(lane, hh);
      int c = k / 9, t9 = k - c * 9;
      int ky = t9 / 3, kx = t9 - ky * 3;
      bf[hh] = sIn[c * 100 + (pi + ky) * 10 + (pj + kx)];
    }
    acc = wmma16(a, bf, acc);
  }
  __syncthreads();
  {
    int n = nt * 16 + (lane & 15);
    int mb = mt * 16 + ((lane & 16) ? 8 : 0);
#pragma unroll
    for (int r = 0; r < 8; ++r) sC1[(mb + r) * 64 + n] = acc[r] + conf1_b[mb + r];
  }
  __syncthreads();
  if (t < 64) {
    float s = conf2_b[0];
#pragma unroll
    for (int o = 0; o < 32; ++o) s += conf2_w[o] * fmaxf(sC1[o * 64 + t], 0.0f);
    float sg = 1.0f / (1.0f + expf(-s));
    int i = t >> 3, j = t & 7;
    confOut[((size_t)b * 128 + ti_ * 8 + i) * 128 + tj_ * 8 + j] = sg;
  }
}

// ---------------------------------------------------------------------------
extern "C" void kernel_launch(void* const* d_in, const int* in_sizes, int n_in,
                              void* d_out, int out_size, void* d_ws, size_t ws_size,
                              hipStream_t stream)
{
  const float* query   = (const float*)d_in[0];
  const float* values  = (const float*)d_in[1];
  const float* rel     = (const float*)d_in[2];
  const float* tgt     = (const float*)d_in[3];
  const float* so1_w   = (const float*)d_in[4];
  const float* so1_b   = (const float*)d_in[5];
  const float* gno_g   = (const float*)d_in[6];
  const float* gno_b   = (const float*)d_in[7];
  const float* so2_w   = (const float*)d_in[8];
  const float* so2_b   = (const float*)d_in[9];
  const float* aw1_w   = (const float*)d_in[10];
  const float* aw1_b   = (const float*)d_in[11];
  const float* gna_g   = (const float*)d_in[12];
  const float* gna_b   = (const float*)d_in[13];
  const float* aw2_w   = (const float*)d_in[14];
  const float* aw2_b   = (const float*)d_in[15];
  const float* vproj_w = (const float*)d_in[16];
  const float* vproj_b = (const float*)d_in[17];
  const float* oproj_w = (const float*)d_in[18];
  const float* oproj_b = (const float*)d_in[19];
  const float* filmo_w = (const float*)d_in[20];
  const float* filmo_b = (const float*)d_in[21];
  const float* filma_w = (const float*)d_in[22];
  const float* filma_b = (const float*)d_in[23];
  const float* dts_w   = (const float*)d_in[24];
  const float* dts_b   = (const float*)d_in[25];
  const float* lvl     = (const float*)d_in[26];
  const float* conf1_w = (const float*)d_in[27];
  const float* conf1_b = (const float*)d_in[28];
  const float* conf2_w = (const float*)d_in[29];
  const float* conf2_b = (const float*)d_in[30];

  char* ws = (char*)d_ws;
  _Float16* wh   = (_Float16*)(ws);                 // 95232 halves = 190464 B
  float* film    = (float*)(ws + 190464);           // 1024 f32 (film_o | film_a)
  float* absdt   = (float*)(ws + 194560);           // 12 f32
  float* hscale  = (float*)(ws + 194624);           // 48 f32
  float* attnW   = (float*)(ws + 194816);           // 1024*64*4*12 f32 = 12.58 MB
  float* coordW  = (float*)(ws + 12777728);         // 2x above = 25.17 MB
  _Float16* vout = (_Float16*)(ws + 37943552);      // 3*1024*128*64 f16 = 50.33 MB

  float* fused = (float*)d_out;                     // (4,128,128,128)
  float* conf  = fused + 8388608;                   // (4,1,128,128)
  float* ent   = conf + 65536;                      // (4,1,128,128)

  prep_kernel<<<377, 256, 0, stream>>>(so1_w, aw1_w, so2_w, aw2_w, vproj_w, oproj_w, conf1_w,
                                       tgt, filmo_w, filmo_b, filma_w, filma_b,
                                       rel, dts_w, dts_b, wh, film, absdt, hscale);
  branch_kernel<<<1024, 256, 0, stream>>>(query, wh, film, absdt, hscale,
                                          so1_b, gno_g, gno_b, so2_b,
                                          aw1_b, gna_g, gna_b, aw2_b,
                                          attnW, coordW, ent);
  vproj_kernel<<<3072, 256, 0, stream>>>(values, lvl, wh, vproj_b, vout);
  sample_oproj_kernel<<<1024, 256, 0, stream>>>(vout, attnW, coordW, wh, oproj_b, fused);
  conf_kernel<<<1024, 256, 0, stream>>>(fused, wh, conf1_b, conf2_w, conf2_b, conf);
}
